// EncoderBlock_75359496176199
// MI455X (gfx1250) — matrix operand order repaired and checked by reading, not compiled, not run on hardware
//
#include <hip/hip_runtime.h>

typedef __bf16 bf16;
typedef __attribute__((ext_vector_type(16))) __bf16 v16bf;
typedef __attribute__((ext_vector_type(8)))  float  v8f;
typedef __attribute__((ext_vector_type(4)))  unsigned int u32x4;

#define D_MODEL 1024
#define ROWS    2048     // B*L
#define NHEADS  32       // B*NB_HEADS under the faithful reshape
#define DHEAD   64
#define SEQ     1024

#define BM 64
#define BN 128
#define BK 32
#define A_TILE_BYTES (BM * BK * 2)          // 4096
#define B_TILE_BYTES (BN * BK * 2)          // 8192
#define BUF_BYTES    (A_TILE_BYTES + B_TILE_BYTES)  // 12288 per buffer

union FragCaster { v16bf v; u32x4 u[2]; };

// ---------------------------------------------------------------------------
// WMMA fragment layouts (wave32, 16x16x32 bf16; CDNA5 ISA 7.12.2)
// A: 16x32 (MxK). lane<16 -> M=lane, K in {0..7}U{16..23}; lane>=16 -> {8..15}U{24..31}
// B: 32x16 (KxN). lane<16 -> N=lane, K in {0..7}U{16..23}; lane>=16 -> N=lane-16, K in {8..15}U{24..31}
// C/D: 16x16 f32.  N = lane&15, M = elem + 8*(lane>>4)
// ---------------------------------------------------------------------------
__device__ __forceinline__ v16bf load_a_frag(const bf16* A, int lda, int mbase,
                                             int k0, int lane) {
  const int m    = mbase + (lane & 15);
  const int half = lane >> 4;
  const bf16* p  = A + (size_t)m * lda + k0 + half * 8;
  FragCaster t;
  t.u[0] = *(const u32x4*)(p);
  t.u[1] = *(const u32x4*)(p + 16);
  return t.v;
}

__device__ __forceinline__ v16bf load_b_frag(const bf16* B, int ldb, int nbase,
                                             int k0, int lane) {
  const int n    = nbase + (lane & 15);
  const int koff = k0 + ((lane >> 4) << 3);
  const bf16* p  = B + (size_t)n * ldb + koff;
  FragCaster t;
  t.u[0] = *(const u32x4*)(p);
  t.u[1] = *(const u32x4*)(p + 16);
  return t.v;
}

// LDS tile variants: tiles are stored K-major with row stride BK(=32) elems.
__device__ __forceinline__ v16bf lds_read_a(const bf16* tile, int mrow, int lane) {
  const int half = lane >> 4;
  const bf16* p  = tile + mrow * BK + half * 8;
  FragCaster t;
  t.u[0] = *(const u32x4*)(p);
  t.u[1] = *(const u32x4*)(p + 16);
  return t.v;
}
__device__ __forceinline__ v16bf lds_read_b(const bf16* tile, int nrow, int lane) {
  const bf16* p = tile + nrow * BK + ((lane >> 4) << 3);
  FragCaster t;
  t.u[0] = *(const u32x4*)(p);
  t.u[1] = *(const u32x4*)(p + 16);
  return t.v;
}

__device__ __forceinline__ v8f wmma_bf16(v16bf a, v16bf b, v8f c) {
  return __builtin_amdgcn_wmma_f32_16x16x32_bf16(false, a, false, b,
                                                 (short)0, c, false, false);
}

__device__ __forceinline__ float wave_max(float v) {
  for (int o = 16; o > 0; o >>= 1) v = fmaxf(v, __shfl_xor(v, o, 32));
  return v;
}
__device__ __forceinline__ float wave_sum(float v) {
  for (int o = 16; o > 0; o >>= 1) v += __shfl_xor(v, o, 32);
  return v;
}

// gfx1250 async global->LDS copy (ASYNCcnt-tracked, no VGPR data path)
__device__ __forceinline__ void async_b128_to_lds(unsigned lds_off, const bf16* g) {
  asm volatile("global_load_async_to_lds_b128 %0, %1, off"
               :: "v"(lds_off), "v"(g) : "memory");
}
__device__ __forceinline__ void wait_async0() {
  asm volatile("s_wait_asynccnt 0x0" ::: "memory");
}

// ---------------------------------------------------------------------------
// fp32 -> bf16 converter
// ---------------------------------------------------------------------------
__global__ void __launch_bounds__(256)
cvt_bf16_kernel(const float* __restrict__ in, bf16* __restrict__ out, int n) {
  int i = blockIdx.x * 256 + threadIdx.x;
  if (i < n) out[i] = (bf16)in[i];
}

// ---------------------------------------------------------------------------
// V transpose: v[head][key][ch] -> vt[head][ch][key]
// ---------------------------------------------------------------------------
__global__ void __launch_bounds__(256)
transpose_v_kernel(const bf16* __restrict__ v, bf16* __restrict__ vt) {
  int idx  = blockIdx.x * 256 + threadIdx.x;   // 0 .. 2M-1
  int head = idx >> 16;
  int rem  = idx & 65535;
  int key  = rem >> 6;
  int ch   = rem & 63;
  vt[(head << 16) + (ch << 10) + key] = v[idx];
}

// ---------------------------------------------------------------------------
// GEMM: out[M,N] = A[M,K] @ Bw[N,K]^T + bias[N] (+ residual)
// Block = 128 threads (4 waves, 2x2). Block tile 64x128, wave tile 32x64.
// A/B tiles double-buffered in LDS, filled with async global->LDS b128 copies
// while the WMMAs consume the other buffer.
// ---------------------------------------------------------------------------
__device__ __forceinline__ void fill_tiles_async(
    unsigned bufBase, const bf16* __restrict__ A, const bf16* __restrict__ Bw,
    int K, int mbB, int nbB, int k0, int tid) {
  // A tile: 64 rows x 64B  = 256 chunks of 16B; 2 per thread
#pragma unroll
  for (int i = 0; i < 2; ++i) {
    int c   = tid + i * 128;
    int row = c >> 2, seg = c & 3;
    async_b128_to_lds(bufBase + row * 64 + seg * 16,
                      A + (size_t)(mbB + row) * K + k0 + seg * 8);
  }
  // B tile: 128 rows x 64B = 512 chunks of 16B; 4 per thread
#pragma unroll
  for (int i = 0; i < 4; ++i) {
    int c   = tid + i * 128;
    int row = c >> 2, seg = c & 3;
    async_b128_to_lds(bufBase + A_TILE_BYTES + row * 64 + seg * 16,
                      Bw + (size_t)(nbB + row) * K + k0 + seg * 8);
  }
}

template <bool RESID, bool OUTF, bool OUTB>
__global__ void __launch_bounds__(128)
gemm_wmma_kernel(const bf16* __restrict__ A, const bf16* __restrict__ Bw,
                 const float* __restrict__ bias, const float* __restrict__ residual,
                 float* __restrict__ outF, bf16* __restrict__ outB,
                 int M, int N, int K) {
  extern __shared__ __align__(16) char smem[];
  const int tid  = threadIdx.x;
  const int lane = tid & 31;
  const int wid  = tid >> 5;
  const int wm   = wid >> 1;
  const int wn   = wid & 1;
  const int mbB  = blockIdx.y * BM;           // block row base
  const int nbB  = blockIdx.x * BN;           // block col base
  const int mb   = mbB + wm * 32;             // wave row base
  const int nb   = nbB + wn * 64;             // wave col base

  v8f acc[2][4] = {};

  fill_tiles_async(0u, A, Bw, K, mbB, nbB, 0, tid);
  unsigned cur = 0;
  for (int k0 = 0; k0 < K; k0 += BK) {
    wait_async0();
    __syncthreads();                           // current buffer ready for all waves
    const int kn = k0 + BK;
    if (kn < K) fill_tiles_async(cur ^ BUF_BYTES, A, Bw, K, mbB, nbB, kn, tid);

    const bf16* tileA = (const bf16*)(smem + cur);
    const bf16* tileB = (const bf16*)(smem + cur + A_TILE_BYTES);
    v16bf a[2], b[4];
#pragma unroll
    for (int i = 0; i < 2; ++i)
      a[i] = lds_read_a(tileA, wm * 32 + i * 16 + (lane & 15), lane);
#pragma unroll
    for (int j = 0; j < 4; ++j)
      b[j] = lds_read_b(tileB, wn * 64 + j * 16 + (lane & 15), lane);
#pragma unroll
    for (int i = 0; i < 2; ++i)
#pragma unroll
      for (int j = 0; j < 4; ++j) acc[i][j] = wmma_bf16(a[i], b[j], acc[i][j]);

    __syncthreads();                           // reads done before buffer is refilled
    cur ^= BUF_BYTES;
  }

  const int half = lane >> 4;
#pragma unroll
  for (int i = 0; i < 2; ++i)
#pragma unroll
    for (int j = 0; j < 4; ++j) {
      const int n = nb + j * 16 + (lane & 15);
      const float bn = bias[n];
#pragma unroll
      for (int e = 0; e < 8; ++e) {
        const int m = mb + i * 16 + half * 8 + e;
        const size_t idx = (size_t)m * N + n;
        float v = acc[i][j][e] + bn;
        if (RESID) v += residual[idx];
        if (OUTF) outF[idx] = v;
        if (OUTB) outB[idx] = (bf16)v;
      }
    }
}

// ---------------------------------------------------------------------------
// Fused attention: per (head, 16-row q-tile), one wave.
//   scores = q_tile @ K^T  (WMMA, kept fp32 in LDS, 16x1024 = 64KB)
//   softmax(row) ; += T5 relative-position bias (post-softmax, per reference)
//   out = scores @ V       (WMMA via pre-transposed V)
// ---------------------------------------------------------------------------
__global__ void __launch_bounds__(32)
attention_kernel(const bf16* __restrict__ q, const bf16* __restrict__ k,
                 const bf16* __restrict__ vt, const float* __restrict__ rel_bias,
                 bf16* __restrict__ att_out) {
  const int qtile = blockIdx.x;   // 0..63
  const int head  = blockIdx.y;   // 0..31
  const int lane  = threadIdx.x;
  const int half  = lane >> 4;

  __shared__ float sc[16 * SEQ];  // 64 KB of the WGP's 320 KB LDS

  const bf16* qh  = q  + (size_t)head * SEQ * DHEAD;
  const bf16* kh  = k  + (size_t)head * SEQ * DHEAD;
  const bf16* vth = vt + (size_t)head * DHEAD * SEQ;

  // ---- scores: M=16(q rows), N=1024(keys), K=64(channels) ----
  v16bf a0 = load_a_frag(qh, DHEAD, qtile * 16, 0, lane);
  v16bf a1 = load_a_frag(qh, DHEAD, qtile * 16, 32, lane);
  for (int kt = 0; kt < SEQ / 16; ++kt) {
    v16bf b0 = load_b_frag(kh, DHEAD, kt * 16, 0, lane);
    v16bf b1 = load_b_frag(kh, DHEAD, kt * 16, 32, lane);
    v8f acc = {};
    acc = wmma_bf16(a0, b0, acc);
    acc = wmma_bf16(a1, b1, acc);
    const int n = lane & 15;
#pragma unroll
    for (int e = 0; e < 8; ++e)
      sc[(half * 8 + e) * SEQ + kt * 16 + n] = acc[e];
  }
  __syncthreads();

  // ---- per-row softmax, then add post-softmax T5 bucket bias ----
  const int h16 = head & 15;
  const float inv_log16 = 8.0f / __logf(16.0f);   // (nb/2 - max_exact)/log(maxd/max_exact)
  for (int r = 0; r < 16; ++r) {
    float* row = sc + r * SEQ;
    float mx = -1e30f;
    for (int j = lane; j < SEQ; j += 32) mx = fmaxf(mx, row[j]);
    mx = wave_max(mx);
    float sum = 0.f;
    for (int j = lane; j < SEQ; j += 32) {
      float e = __expf(row[j] - mx);
      row[j] = e;
      sum += e;
    }
    sum = wave_sum(sum);
    const float inv = 1.0f / sum;
    const int qi = qtile * 16 + r;
    for (int j = lane; j < SEQ; j += 32) {
      int nn  = j - qi;                       // n = -(q - k)
      int ret = (nn < 0) ? 16 : 0;            // bidirectional half
      int na  = nn < 0 ? -nn : nn;
      int bkt;
      if (na < 8) bkt = na;
      else {
        int vl = 8 + (int)(__logf((float)na * 0.125f) * inv_log16);
        bkt = vl < 15 ? vl : 15;
      }
      row[j] = row[j] * inv + rel_bias[(ret + bkt) * 16 + h16];
    }
  }
  __syncthreads();

  // ---- out = P @ V : M=16, N=64(channels), K=1024(keys) ----
  v8f acc[4] = {};
  const int mrow = lane & 15;
  for (int ks = 0; ks < SEQ / 32; ++ks) {
    v16bf a;
#pragma unroll
    for (int e = 0; e < 8; ++e) {
      a[e]     = (bf16)sc[mrow * SEQ + ks * 32 + half * 8 + e];
      a[e + 8] = (bf16)sc[mrow * SEQ + ks * 32 + 16 + half * 8 + e];
    }
#pragma unroll
    for (int ct = 0; ct < 4; ++ct) {
      v16bf b = load_b_frag(vth, SEQ, ct * 16, ks * 32, lane);
      acc[ct] = wmma_bf16(a, b, acc[ct]);
    }
  }
#pragma unroll
  for (int ct = 0; ct < 4; ++ct) {
    const int n = ct * 16 + (lane & 15);
#pragma unroll
    for (int e = 0; e < 8; ++e) {
      const int m = half * 8 + e;
      att_out[(size_t)head * (SEQ * DHEAD) + (size_t)(qtile * 16 + m) * DHEAD + n] =
          (bf16)acc[ct][e];
    }
  }
}

// ---------------------------------------------------------------------------
// Eps-free layernorm over rows of 1024 (biased std), matching the reference
// ---------------------------------------------------------------------------
__global__ void __launch_bounds__(256)
layernorm_kernel(const float* __restrict__ in, float* __restrict__ outF,
                 bf16* __restrict__ outB) {
  const int row = blockIdx.x;
  const float* p = in + (size_t)row * D_MODEL;
  float vals[4];
  float s = 0.f, sq = 0.f;
#pragma unroll
  for (int i = 0; i < 4; ++i) {
    float v = p[threadIdx.x + i * 256];
    vals[i] = v;
    s += v;
    sq += v * v;
  }
  __shared__ float ssum[8], ssq[8];
  const int lane = threadIdx.x & 31, wid = threadIdx.x >> 5;
  s = wave_sum(s);
  sq = wave_sum(sq);
  if (lane == 0) { ssum[wid] = s; ssq[wid] = sq; }
  __syncthreads();
  if (wid == 0) {
    float a = (lane < 8) ? ssum[lane] : 0.f;
    float b = (lane < 8) ? ssq[lane] : 0.f;
    for (int o = 4; o > 0; o >>= 1) { a += __shfl_xor(a, o, 32); b += __shfl_xor(b, o, 32); }
    if (lane == 0) { ssum[0] = a; ssq[0] = b; }
  }
  __syncthreads();
  const float mean = ssum[0] * (1.0f / D_MODEL);
  const float var  = ssq[0] * (1.0f / D_MODEL) - mean * mean;
  const float inv  = 1.0f / __fsqrt_rn(var > 0.f ? var : 1e-30f);
#pragma unroll
  for (int i = 0; i < 4; ++i) {
    const size_t idx = (size_t)row * D_MODEL + threadIdx.x + i * 256;
    float o = (vals[i] - mean) * inv;
    if (outF) outF[idx] = o;
    if (outB) outB[idx] = (bf16)o;
  }
}

// ---------------------------------------------------------------------------
extern "C" void kernel_launch(void* const* d_in, const int* in_sizes, int n_in,
                              void* d_out, int out_size, void* d_ws, size_t ws_size,
                              hipStream_t stream) {
  const float* x        = (const float*)d_in[0];
  const float* Wq       = (const float*)d_in[1];
  const float* bq       = (const float*)d_in[2];
  const float* Wk       = (const float*)d_in[3];
  const float* bk       = (const float*)d_in[4];
  const float* Wv       = (const float*)d_in[5];
  const float* bv       = (const float*)d_in[6];
  const float* Wo       = (const float*)d_in[7];
  const float* bo       = (const float*)d_in[8];
  const float* rel_bias = (const float*)d_in[9];
  const float* W1       = (const float*)d_in[10];
  const float* b1       = (const float*)d_in[11];
  const float* W2       = (const float*)d_in[12];
  const float* b2       = (const float*)d_in[13];

  char* ws = (char*)d_ws;
  size_t off = 0;
  auto alloc = [&](size_t bytes) -> void* {
    void* p = ws + off;
    off += (bytes + 255) & ~(size_t)255;
    return p;
  };
  const size_t NTOK = (size_t)ROWS * D_MODEL;      // 2M elements
  const size_t NW   = (size_t)D_MODEL * D_MODEL;   // 1M elements

  bf16* xb   = (bf16*)alloc(NTOK * 2);
  bf16* Wqb  = (bf16*)alloc(NW * 2);
  bf16* Wkb  = (bf16*)alloc(NW * 2);
  bf16* Wvb  = (bf16*)alloc(NW * 2);
  bf16* Wob  = (bf16*)alloc(NW * 2);
  bf16* W1b  = (bf16*)alloc(NW * 2);
  bf16* W2b  = (bf16*)alloc(NW * 2);
  bf16* qb   = (bf16*)alloc(NTOK * 2);
  bf16* kb   = (bf16*)alloc(NTOK * 2);
  bf16* vb   = (bf16*)alloc(NTOK * 2);
  bf16* vtb  = (bf16*)alloc(NTOK * 2);
  bf16* attb = (bf16*)alloc(NTOK * 2);
  float* attn_sum   = (float*)alloc(NTOK * 4);
  float* attn_block = (float*)alloc(NTOK * 4);
  bf16*  attn_blk_b = (bf16*)alloc(NTOK * 2);
  bf16*  h1         = (bf16*)alloc(NTOK * 2);
  float* pre2       = (float*)alloc(NTOK * 4);

  // 1) fp32 -> bf16 conversions
  cvt_bf16_kernel<<<NTOK / 256, 256, 0, stream>>>(x, xb, (int)NTOK);
  cvt_bf16_kernel<<<NW / 256, 256, 0, stream>>>(Wq, Wqb, (int)NW);
  cvt_bf16_kernel<<<NW / 256, 256, 0, stream>>>(Wk, Wkb, (int)NW);
  cvt_bf16_kernel<<<NW / 256, 256, 0, stream>>>(Wv, Wvb, (int)NW);
  cvt_bf16_kernel<<<NW / 256, 256, 0, stream>>>(Wo, Wob, (int)NW);
  cvt_bf16_kernel<<<NW / 256, 256, 0, stream>>>(W1, W1b, (int)NW);
  cvt_bf16_kernel<<<NW / 256, 256, 0, stream>>>(W2, W2b, (int)NW);

  const int LDSB = 2 * BUF_BYTES;   // 24 KB double-buffered tiles
  dim3 gg(D_MODEL / BN, ROWS / BM);

  // 2) Q,K,V projections (bf16 out for attention)
  gemm_wmma_kernel<false, false, true><<<gg, 128, LDSB, stream>>>(
      xb, Wqb, bq, nullptr, nullptr, qb, ROWS, D_MODEL, D_MODEL);
  gemm_wmma_kernel<false, false, true><<<gg, 128, LDSB, stream>>>(
      xb, Wkb, bk, nullptr, nullptr, kb, ROWS, D_MODEL, D_MODEL);
  gemm_wmma_kernel<false, false, true><<<gg, 128, LDSB, stream>>>(
      xb, Wvb, bv, nullptr, nullptr, vb, ROWS, D_MODEL, D_MODEL);

  // 3) transpose V for K-contiguous PV fragments
  transpose_v_kernel<<<NTOK / 256, 256, 0, stream>>>(vb, vtb);

  // 4) fused attention (QK^T -> softmax -> +bias -> PV), per head x q-tile
  dim3 ga(SEQ / 16, NHEADS);
  attention_kernel<<<ga, 32, 0, stream>>>(qb, kb, vtb, rel_bias, attb);

  // 5) output projection + residual x
  gemm_wmma_kernel<true, true, false><<<gg, 128, LDSB, stream>>>(
      attb, Wob, bo, x, attn_sum, nullptr, ROWS, D_MODEL, D_MODEL);

  // 6) layernorm 1 (fp32 + bf16 copy for FFN)
  layernorm_kernel<<<ROWS, 256, 0, stream>>>(attn_sum, attn_block, attn_blk_b);

  // 7) FFN (no activation, per reference)
  gemm_wmma_kernel<false, false, true><<<gg, 128, LDSB, stream>>>(
      attn_blk_b, W1b, b1, nullptr, nullptr, h1, ROWS, D_MODEL, D_MODEL);
  gemm_wmma_kernel<true, true, false><<<gg, 128, LDSB, stream>>>(
      h1, W2b, b2, attn_block, pre2, nullptr, ROWS, D_MODEL, D_MODEL);

  // 8) layernorm 2 -> final output
  layernorm_kernel<<<ROWS, 256, 0, stream>>>(pre2, (float*)d_out, nullptr);
}
